// FusedGCNLayer_35210141893096
// MI455X (gfx1250) — matrix operand order repaired and checked by reading, not compiled, not run on hardware
//
#include <hip/hip_runtime.h>

// ---------------------------------------------------------------------------
// FusedGCNLayer for gfx1250 (MI455X):
//   1) convert_w : W(fp32) -> W_hi/W_lo (bf16 split), 0.5MB, L2-resident
//   2) gemm      : h = x @ W^T via bf16x3 split on v_wmma_f32_16x16x32_bf16
//                  32-row M-block -> each wave reuses its B fragment for two
//                  16x16 output tiles; x staged to LDS as packed bf16 pairs
//                  (global b64 -> v_cvt_pk_bf16_f32 -> ds_store_b32/2addr)
//   3) zero_out  : out = 0 (b128 stores)
//   4) scatter   : out[row[e]] += vals[e] * h[col[e]]  (global_atomic_add_f32)
// ---------------------------------------------------------------------------

#define FEAT 1024
#define EMB  128
#define MROWS 32      // rows of x per block (two 16-row WMMA tiles)
#define XS_PITCHW 20  // row pitch in dwords (16 payload + 4 pad) -> conflict-free b128 reads

typedef __attribute__((ext_vector_type(16))) __bf16 v16bf;
typedef __attribute__((ext_vector_type(8)))  float  v8f;
typedef __attribute__((ext_vector_type(2)))  float  v2f;
typedef __attribute__((ext_vector_type(2)))  __bf16 v2bf;

union FragAB {
    v16bf v;
    uint4 q[2];   // 32 bytes == 16 x bf16
};

// float2 -> packed bf16 pair (one v_cvt_pk_bf16_f32), plus packed bf16 residual pair
static __device__ __forceinline__ void split2(v2f f, unsigned int& hi_pk, unsigned int& lo_pk) {
    v2bf hb = __builtin_convertvector(f, v2bf);          // v_cvt_pk_bf16_f32 (both slots)
    v2f  hf = __builtin_convertvector(hb, v2f);          // exact up-convert
    v2bf lb = __builtin_convertvector(f - hf, v2bf);     // residual, RNE
    hi_pk = __builtin_bit_cast(unsigned int, hb);
    lo_pk = __builtin_bit_cast(unsigned int, lb);
}

// ---- 1) split W into bf16 hi/lo (packed dword writes) ----------------------
__global__ __launch_bounds__(256) void gcn_convert_w(const v2f* __restrict__ W2,
                                                     unsigned int* __restrict__ whi,
                                                     unsigned int* __restrict__ wlo,
                                                     int n2) {
    int i = blockIdx.x * 256 + threadIdx.x;
    if (i < n2) {
        unsigned int hp, lp;
        split2(W2[i], hp, lp);
        whi[i] = hp;
        wlo[i] = lp;
    }
}

// ---- 2) dense GEMM: h[n,128] = x[n,1024] @ W^T  (bf16x3 on WMMA) -----------
// Block = 256 threads = 8 waves. Block computes a 32-row x 128-col stripe of h;
// wave w owns columns [16w, 16w+16) for BOTH 16-row tiles (B-fragment reuse).
__global__ __launch_bounds__(256) void gcn_gemm_bf16x3(const float* __restrict__ x,
                                                       const unsigned short* __restrict__ whi,
                                                       const unsigned short* __restrict__ wlo,
                                                       float* __restrict__ h,
                                                       int n_nodes) {
    __shared__ __align__(16) unsigned int xs_hi[MROWS][XS_PITCHW];
    __shared__ __align__(16) unsigned int xs_lo[MROWS][XS_PITCHW];

    const int tid    = threadIdx.x;
    const int lane   = tid & 31;
    const int wave   = tid >> 5;          // N-tile index 0..7
    const int m0     = blockIdx.x * MROWS;
    const int n0     = wave * 16;
    const int mA     = lane & 15;         // row (A) / column (B,C) within tile
    const int hiHalf = lane >> 4;         // 0: lanes 0-15, 1: lanes 16-31

    v8f acc0 = {0.f, 0.f, 0.f, 0.f, 0.f, 0.f, 0.f, 0.f};
    v8f acc1 = acc0;

    const int last = n_nodes - 1;
    const int kaW  = hiHalf * 4;          // A fragment dword offset within row (0 or 4)

    // ushort views for b128 fragment reads
    const unsigned short* xs_hi_us = (const unsigned short*)xs_hi;
    const unsigned short* xs_lo_us = (const unsigned short*)xs_lo;

    for (int k0 = 0; k0 < FEAT; k0 += 32) {
        // --- stage one 32x32 fp32 chunk of x into LDS as packed bf16 hi/lo ---
        // 512 float2's; thread handles 2: coalesced b64 loads, packed b32 stores.
        #pragma unroll
        for (int j = 0; j < 2; ++j) {
            int p  = tid + j * 256;       // float2 index over 32 rows x 16 pairs
            int r  = p >> 4;
            int c2 = p & 15;
            int row = m0 + r; row = row > last ? last : row;
            v2f f2 = *(const v2f*)&x[(size_t)row * FEAT + k0 + 2 * c2];
            unsigned int hp, lp;
            split2(f2, hp, lp);
            xs_hi[r][c2] = hp;            // ds_store_b32 (pairs -> 2addr forms)
            xs_lo[r][c2] = lp;
        }
        __syncthreads();

        // --- B fragments from global (L2/WGP$-resident), reused by both M tiles ---
        // 16-bit B 32x16 layout (matches A): elems 0-7 -> K 8h..8h+7, 8-15 -> K 16+8h..
        // two b128 reads per fragment.
        const size_t boff = (size_t)(n0 + mA) * FEAT + (size_t)k0 + hiHalf * 8;
        FragAB bhi, blo;
        const uint4* ph = (const uint4*)(whi + boff);
        const uint4* pl = (const uint4*)(wlo + boff);
        bhi.q[0] = ph[0]; bhi.q[1] = ph[2];
        blo.q[0] = pl[0]; blo.q[1] = pl[2];

        // --- A fragments from LDS (ISA 16-bit A 16x32 layout), tile 0 then tile 1 ---
        // lane<16: elems 0-7 -> K 0-7, 8-15 -> K 16-23 ; lane>=16: +8
        const int rowBase0 = mA * (2 * XS_PITCHW);        // ushort index of row mA
        FragAB ahi, alo;
        ahi.q[0] = *(const uint4*)&xs_hi_us[rowBase0 + 2 * kaW];
        ahi.q[1] = *(const uint4*)&xs_hi_us[rowBase0 + 2 * kaW + 16];
        alo.q[0] = *(const uint4*)&xs_lo_us[rowBase0 + 2 * kaW];
        alo.q[1] = *(const uint4*)&xs_lo_us[rowBase0 + 2 * kaW + 16];
        acc0 = __builtin_amdgcn_wmma_f32_16x16x32_bf16(false, ahi.v, false, bhi.v,
                                                       (short)0, acc0, false, false);
        acc0 = __builtin_amdgcn_wmma_f32_16x16x32_bf16(false, ahi.v, false, blo.v,
                                                       (short)0, acc0, false, false);
        acc0 = __builtin_amdgcn_wmma_f32_16x16x32_bf16(false, alo.v, false, bhi.v,
                                                       (short)0, acc0, false, false);

        const int rowBase1 = (16 + mA) * (2 * XS_PITCHW);
        ahi.q[0] = *(const uint4*)&xs_hi_us[rowBase1 + 2 * kaW];
        ahi.q[1] = *(const uint4*)&xs_hi_us[rowBase1 + 2 * kaW + 16];
        alo.q[0] = *(const uint4*)&xs_lo_us[rowBase1 + 2 * kaW];
        alo.q[1] = *(const uint4*)&xs_lo_us[rowBase1 + 2 * kaW + 16];
        acc1 = __builtin_amdgcn_wmma_f32_16x16x32_bf16(false, ahi.v, false, bhi.v,
                                                       (short)0, acc1, false, false);
        acc1 = __builtin_amdgcn_wmma_f32_16x16x32_bf16(false, ahi.v, false, blo.v,
                                                       (short)0, acc1, false, false);
        acc1 = __builtin_amdgcn_wmma_f32_16x16x32_bf16(false, alo.v, false, bhi.v,
                                                       (short)0, acc1, false, false);
        __syncthreads();
    }

    // --- store C (VGPR v: lanes 0-15 -> M=v, lanes 16-31 -> M=v+8; N=lane%16) ---
    #pragma unroll
    for (int v = 0; v < 8; ++v) {
        int m = m0 + v + hiHalf * 8;
        if (m < n_nodes)      h[(size_t)m * EMB + n0 + mA]        = acc0[v];
        if (m + 16 < n_nodes) h[(size_t)(m + 16) * EMB + n0 + mA] = acc1[v];
    }
}

// ---- 3) zero the output (harness poisons d_out) ----------------------------
__global__ __launch_bounds__(256) void gcn_zero(float4* __restrict__ out, int n4) {
    int stride = gridDim.x * 256;
    float4 z = {0.f, 0.f, 0.f, 0.f};
    for (int i = blockIdx.x * 256 + threadIdx.x; i < n4; i += stride) out[i] = z;
}

// ---- 4) edge scatter: one wave per edge, one b128 load + 4 f32 atomics/lane -
__global__ __launch_bounds__(256) void gcn_scatter(const int* __restrict__ row,
                                                   const int* __restrict__ col,
                                                   const float* __restrict__ vals,
                                                   const float* __restrict__ h,
                                                   float* __restrict__ out,
                                                   int n_edges) {
    int e = blockIdx.x * 8 + (threadIdx.x >> 5);
    if (e >= n_edges) return;
    int lane = threadIdx.x & 31;
    int dst = row[e];
    int src = col[e];
    float v = vals[e];
    const float4* hv = (const float4*)(h + (size_t)src * EMB);   // 512B row, L2-resident
    float4 m = hv[lane];                                         // global_load_b128
    float* op = out + (size_t)dst * EMB + lane * 4;
    (void)__hip_atomic_fetch_add(op + 0, v * m.x, __ATOMIC_RELAXED, __HIP_MEMORY_SCOPE_AGENT);
    (void)__hip_atomic_fetch_add(op + 1, v * m.y, __ATOMIC_RELAXED, __HIP_MEMORY_SCOPE_AGENT);
    (void)__hip_atomic_fetch_add(op + 2, v * m.z, __ATOMIC_RELAXED, __HIP_MEMORY_SCOPE_AGENT);
    (void)__hip_atomic_fetch_add(op + 3, v * m.w, __ATOMIC_RELAXED, __HIP_MEMORY_SCOPE_AGENT);
}

// ---------------------------------------------------------------------------
extern "C" void kernel_launch(void* const* d_in, const int* in_sizes, int n_in,
                              void* d_out, int out_size, void* d_ws, size_t ws_size,
                              hipStream_t stream) {
    const float* x     = (const float*)d_in[0];
    const float* W     = (const float*)d_in[1];
    const int*   arow  = (const int*)d_in[2];
    const int*   acol  = (const int*)d_in[3];
    const float* avals = (const float*)d_in[4];

    const int n_nodes = in_sizes[0] / FEAT;
    const int n_w     = in_sizes[1];          // 128*1024
    const int n_edges = in_sizes[2];

    char* ws = (char*)d_ws;
    unsigned short* whi = (unsigned short*)ws;
    unsigned short* wlo = (unsigned short*)(ws + (size_t)n_w * 2);
    float*          h   = (float*)(ws + (size_t)n_w * 4);
    float*          out = (float*)d_out;

    // 1) split W once (0.5MB -> two 0.25MB bf16 planes, stays in L2)
    gcn_convert_w<<<(n_w / 2 + 255) / 256, 256, 0, stream>>>(
        (const v2f*)W, (unsigned int*)whi, (unsigned int*)wlo, n_w / 2);

    // 2) dense GEMM on the WMMA pipe
    int gblocks = (n_nodes + MROWS - 1) / MROWS;
    gcn_gemm_bf16x3<<<gblocks, 256, 0, stream>>>(x, whi, wlo, h, n_nodes);

    // 3) zero output
    gcn_zero<<<1024, 256, 0, stream>>>((float4*)out, out_size / 4);

    // 4) scatter-add over edges (h and out are L2-resident)
    gcn_scatter<<<(n_edges + 7) / 8, 256, 0, stream>>>(arow, acol, avals, h, out, n_edges);
}
